// MultiGPUMoE_6150393168449
// MI455X (gfx1250) — compile-verified, matrix operand order checked by reading, not run on hardware
//
#include <hip/hip_runtime.h>
#include <hip/hip_bf16.h>
#include <math.h>

// Problem constants (match reference)
#define BB 4
#define SS 2048
#define HH 1024
#define EE 8
#define FF 4096
#define TT (BB * SS)   // 8192 tokens

// MoE tile config
#define M_TILE 64          // tokens per block (4 M-groups of 16)
#define FB 256             // F-chunk width (16 N-tiles -> 1 per wave)
#define BLOCK 512          // 16 waves
#define XP 1032            // x LDS row pitch (bf16), pad 8 -> rows 4 banks apart
#define AP 264             // act LDS row pitch (bf16), pad 8

// Workspace layout (bytes)
#define W1T_OFF   0ull
#define W1T_BYTES ((size_t)EE * FF * HH * 2)            // 64 MB  bf16 [E][F][H]
#define W2T_OFF   (W1T_OFF + W1T_BYTES)
#define W2T_BYTES ((size_t)EE * FF * HH * 2)            // 64 MB  bf16 [E][H][F]
#define IDXA_OFF  (W2T_OFF + W2T_BYTES)                 // int [T]
#define IDXB_OFF  (IDXA_OFF + (size_t)TT * 4)
#define PA_OFF    (IDXB_OFF + (size_t)TT * 4)           // float [T]
#define PB_OFF    (PA_OFF + (size_t)TT * 4)
#define CNT_OFF   (PB_OFF + (size_t)TT * 4)             // int [E] (padded)
#define TOK_OFF   (CNT_OFF + 256)                       // int [E][T]
#define PRB_OFF   (TOK_OFF + (size_t)EE * TT * 4)       // float [E][T]

typedef __attribute__((ext_vector_type(16))) __bf16 v16bf;
typedef __attribute__((ext_vector_type(8)))  float  v8f;

union F16x16 { v16bf v; uint4 q[2]; };

__device__ inline v16bf ld_frag(const __bf16* p0, const __bf16* p1) {
    F16x16 f;
    f.q[0] = *reinterpret_cast<const uint4*>(p0);
    f.q[1] = *reinterpret_cast<const uint4*>(p1);
    return f.v;
}

__device__ inline v8f wmma_bf16(v16bf a, v16bf b, v8f c) {
    return __builtin_amdgcn_wmma_f32_16x16x32_bf16(
        false, a, false, b, (short)0, c, false, false);
}

// gelu(tanh approx): 0.5x(1+tanh(z)) == x * sigmoid(2z),  z = c0*(x + 0.044715 x^3)
__device__ inline float gelu_tanh(float x) {
    const float c2 = 1.5957691216057308f;   // 2*sqrt(2/pi)
    float z2 = c2 * (x + 0.044715f * x * x * x);
    return x * (1.0f / (1.0f + __expf(-z2)));
}

// ---------------------------------------------------------------------------
// 1) Transpose + convert weights to bf16.
//    w1 [E][H][F] f32 -> w1t [E][F][H] bf16
//    w2 [E][F][H] f32 -> w2t [E][H][F] bf16
// ---------------------------------------------------------------------------
__global__ void transpose_convert_kernel(const float* __restrict__ w1,
                                         const float* __restrict__ w2,
                                         char* __restrict__ ws) {
    __shared__ float t[32][33];
    const int tid = threadIdx.x;
    const int c = tid & 31, r8 = tid >> 5;

    int b = blockIdx.x;
    const float* inM;
    __bf16* outM;
    int inCols, outCols, r0, c0;
    const int tilesW1 = EE * (HH / 32) * (FF / 32);     // 32768

    if (b < tilesW1) {
        int e  = b / ((HH / 32) * (FF / 32));
        int rm = b % ((HH / 32) * (FF / 32));
        int ht = rm / (FF / 32);
        int ft = rm % (FF / 32);
        inM  = w1 + (size_t)e * HH * FF;
        outM = (__bf16*)(ws + W1T_OFF) + (size_t)e * FF * HH;
        inCols = FF; outCols = HH;
        r0 = ht * 32; c0 = ft * 32;
    } else {
        int b2 = b - tilesW1;
        int e  = b2 / ((FF / 32) * (HH / 32));
        int rm = b2 % ((FF / 32) * (HH / 32));
        int ft = rm / (HH / 32);
        int ht = rm % (HH / 32);
        inM  = w2 + (size_t)e * FF * HH;
        outM = (__bf16*)(ws + W2T_OFF) + (size_t)e * HH * FF;
        inCols = HH; outCols = FF;
        r0 = ft * 32; c0 = ht * 32;
    }

#pragma unroll
    for (int i = 0; i < 4; i++) {
        int r = r8 + i * 8;
        t[r][c] = inM[(size_t)(r0 + r) * inCols + c0 + c];
    }
    __syncthreads();
#pragma unroll
    for (int i = 0; i < 4; i++) {
        int r = r8 + i * 8;
        outM[(size_t)(c0 + r) * outCols + r0 + c] = (__bf16)t[c][r];
    }
}

// ---------------------------------------------------------------------------
// 2) Router: per-token top-2 + softmax over the two scores.
// ---------------------------------------------------------------------------
__global__ void router_kernel(const float* __restrict__ logits, char* __restrict__ ws) {
    int t = blockIdx.x * blockDim.x + threadIdx.x;
    if (t >= TT) return;
    float l[EE];
#pragma unroll
    for (int e = 0; e < EE; e++) l[e] = logits[(size_t)t * EE + e];

    int i0 = 0; float s0 = l[0];
#pragma unroll
    for (int e = 1; e < EE; e++) { if (l[e] > s0) { s0 = l[e]; i0 = e; } }
    int i1 = (i0 == 0) ? 1 : 0; float s1 = l[i1];
#pragma unroll
    for (int e = 0; e < EE; e++) { if (e != i0 && l[e] > s1) { s1 = l[e]; i1 = e; } }

    float e1  = __expf(s1 - s0);
    float inv = 1.0f / (1.0f + e1);

    ((int*)  (ws + IDXA_OFF))[t] = i0;
    ((int*)  (ws + IDXB_OFF))[t] = i1;
    ((float*)(ws + PA_OFF ))[t] = inv;
    ((float*)(ws + PB_OFF ))[t] = e1 * inv;
}

// ---------------------------------------------------------------------------
// 3) Deterministic per-expert compaction: one wave32 per expert.
// ---------------------------------------------------------------------------
__global__ void compact_kernel(char* __restrict__ ws) {
    const int lane = threadIdx.x & 31;
    const int e    = threadIdx.x >> 5;          // 8 waves = 8 experts
    const int*   idxA = (const int*)  (ws + IDXA_OFF);
    const int*   idxB = (const int*)  (ws + IDXB_OFF);
    const float* pA   = (const float*)(ws + PA_OFF);
    const float* pB   = (const float*)(ws + PB_OFF);
    int*   toks = (int*)  (ws + TOK_OFF) + (size_t)e * TT;
    float* prbs = (float*)(ws + PRB_OFF) + (size_t)e * TT;

    int base = 0;
    for (int t0 = 0; t0 < TT; t0 += 32) {
        int t = t0 + lane;
        int a = idxA[t], b = idxB[t];
        bool  f = (a == e) || (b == e);
        float p = (a == e) ? pA[t] : ((b == e) ? pB[t] : 0.0f);
        unsigned long long mask = __ballot(f);
        int prefix = __popcll(mask & ((1ull << lane) - 1ull));
        if (f) { toks[base + prefix] = t; prbs[base + prefix] = p; }
        base += __popcll(mask);
    }
    if (lane == 0) ((int*)(ws + CNT_OFF))[e] = base;
}

// ---------------------------------------------------------------------------
// 4) Zero the output (harness poisons it).
// ---------------------------------------------------------------------------
__global__ void zero_out_kernel(float4* __restrict__ out, int n4) {
    int i = blockIdx.x * blockDim.x + threadIdx.x;
    int stride = gridDim.x * blockDim.x;
    float4 z = make_float4(0.f, 0.f, 0.f, 0.f);
    for (; i < n4; i += stride) out[i] = z;
}

// ---------------------------------------------------------------------------
// 5) Fused expert FFN: out[t] += p * ( gelu(x[t] @ w1e) @ w2e )
//    Block: 64 routed tokens of one expert, 16 waves (wave32).
//    GEMM1: wave w owns N-tile w of the 256-wide F-chunk, all 4 M-groups.
//    GEMM2: wave w owns 64 H-columns, all 4 M-groups.
//    Every weight B-fragment feeds 4 WMMAs; acc = 128 VGPRs/wave.
// ---------------------------------------------------------------------------
__global__ void __launch_bounds__(BLOCK)
moe_ffn_kernel(const float* __restrict__ x, const char* __restrict__ ws,
               float* __restrict__ out) {
    const int e    = blockIdx.x >> 7;       // 128 tiles per expert
    const int tile = blockIdx.x & 127;
    const int cnt  = ((const int*)(ws + CNT_OFF))[e];
    if (tile * M_TILE >= cnt) return;       // uniform exit

    const __bf16* w1t  = (const __bf16*)(ws + W1T_OFF) + (size_t)e * FF * HH;
    const __bf16* w2t  = (const __bf16*)(ws + W2T_OFF) + (size_t)e * HH * FF;
    const int*    toks = (const int*)  (ws + TOK_OFF) + (size_t)e * TT;
    const float*  prbs = (const float*)(ws + PRB_OFF) + (size_t)e * TT;

    extern __shared__ char smem[];
    __bf16* xs   = (__bf16*)smem;                         // [64][XP]
    __bf16* as   = xs + M_TILE * XP;                      // [64][AP]
    int*    stok = (int*)(as + M_TILE * AP);              // [64]
    float*  sp   = (float*)(stok + M_TILE);               // [64]

    const int tid  = threadIdx.x;
    const int lane = tid & 31;
    const int w    = tid >> 5;       // wave id 0..15
    const int half = lane >> 4;
    const int klo  = half * 8;
    const int ln   = lane & 15;

    // Stage token indices + probs
    if (tid < M_TILE) {
        int row = tile * M_TILE + tid;
        bool v  = row < cnt;
        stok[tid] = v ? toks[row] : toks[0];
        sp[tid]   = v ? prbs[row] : 0.0f;
    }
    __syncthreads();

    // Stage x tile -> LDS bf16 (coalesced float4 reads)
    for (int j = tid; j < M_TILE * (HH / 4); j += BLOCK) {
        int m  = j >> 8;
        int k4 = (j & 255) << 2;
        float4 v = *(const float4*)(x + (size_t)stok[m] * HH + k4);
        __bf16* d = xs + m * XP + k4;
        d[0] = (__bf16)v.x; d[1] = (__bf16)v.y;
        d[2] = (__bf16)v.z; d[3] = (__bf16)v.w;
    }
    __syncthreads();

    v8f acc2[4][4];                  // [M-group][N-tile] -> 128 VGPRs
#pragma unroll
    for (int mg = 0; mg < 4; mg++)
#pragma unroll
        for (int nt = 0; nt < 4; nt++) acc2[mg][nt] = (v8f)(0.0f);

    // Per-lane A row pointers for the 4 M-groups (GEMM1 A, from LDS x-tile)
    const __bf16* arow0 = xs + (0 * 16 + ln) * XP;
    const __bf16* arow1 = xs + (1 * 16 + ln) * XP;
    const __bf16* arow2 = xs + (2 * 16 + ln) * XP;
    const __bf16* arow3 = xs + (3 * 16 + ln) * XP;

    // GEMM1 B base row: w1t is [F][H], K=H contiguous; wave's N-tile = w
    const __bf16* b1base = w1t + ((size_t)(w * 16 + ln)) * HH;
    // GEMM2 B base row: w2t is [H][F], K=F contiguous; wave owns cols w*64..+64
    const __bf16* b2base = w2t + ((size_t)(w * 64 + ln)) * FF;

#pragma unroll 1
    for (int fc = 0; fc < FF / FB; fc++) {
        const int f0 = fc * FB;

        // Prefetch next chunk's weight rows into the cache hierarchy
        if (fc + 1 < FF / FB) {
            __builtin_prefetch(b1base + (size_t)(f0 + FB) * HH, 0, 1);
            __builtin_prefetch(b2base + (f0 + FB), 0, 1);
        }

        // ---- GEMM1: act[64,256] = x[64,1024] @ w1e[1024,256] ----
        v8f c1[4];
#pragma unroll
        for (int mg = 0; mg < 4; mg++) c1[mg] = (v8f)(0.0f);

        const __bf16* b1p = b1base + (size_t)f0 * HH;
#pragma unroll 4
        for (int k = 0; k < HH; k += 32) {
            v16bf B  = ld_frag(b1p + k + klo, b1p + k + 16 + klo);
            v16bf A0 = ld_frag(arow0 + k + klo, arow0 + k + 16 + klo);
            v16bf A1 = ld_frag(arow1 + k + klo, arow1 + k + 16 + klo);
            v16bf A2 = ld_frag(arow2 + k + klo, arow2 + k + 16 + klo);
            v16bf A3 = ld_frag(arow3 + k + klo, arow3 + k + 16 + klo);
            c1[0] = wmma_bf16(A0, B, c1[0]);
            c1[1] = wmma_bf16(A1, B, c1[1]);
            c1[2] = wmma_bf16(A2, B, c1[2]);
            c1[3] = wmma_bf16(A3, B, c1[3]);
        }
        // gelu + pack act tile to LDS (C layout: VGPR r -> M = r + 8*half)
#pragma unroll
        for (int mg = 0; mg < 4; mg++) {
#pragma unroll
            for (int r = 0; r < 8; r++) {
                int m = mg * 16 + half * 8 + r;
                as[m * AP + w * 16 + ln] = (__bf16)gelu_tanh(c1[mg][r]);
            }
        }
        __syncthreads();

        // ---- GEMM2: out[64,1024] += act[64,256] @ w2e[256,1024] ----
#pragma unroll 2
        for (int ks = 0; ks < FB / 32; ks++) {
            const int kk = ks * 32;
            v16bf A20 = ld_frag(as + (0 * 16 + ln) * AP + kk + klo,
                                as + (0 * 16 + ln) * AP + kk + 16 + klo);
            v16bf A21 = ld_frag(as + (1 * 16 + ln) * AP + kk + klo,
                                as + (1 * 16 + ln) * AP + kk + 16 + klo);
            v16bf A22 = ld_frag(as + (2 * 16 + ln) * AP + kk + klo,
                                as + (2 * 16 + ln) * AP + kk + 16 + klo);
            v16bf A23 = ld_frag(as + (3 * 16 + ln) * AP + kk + klo,
                                as + (3 * 16 + ln) * AP + kk + 16 + klo);
#pragma unroll
            for (int nt = 0; nt < 4; nt++) {
                const __bf16* b2p = b2base + (size_t)(nt * 16) * FF + f0 + kk;
                v16bf B2 = ld_frag(b2p + klo, b2p + 16 + klo);
                acc2[0][nt] = wmma_bf16(A20, B2, acc2[0][nt]);
                acc2[1][nt] = wmma_bf16(A21, B2, acc2[1][nt]);
                acc2[2][nt] = wmma_bf16(A22, B2, acc2[2][nt]);
                acc2[3][nt] = wmma_bf16(A23, B2, acc2[3][nt]);
            }
        }
        __syncthreads();
    }

    // ---- scale by routing prob, combine via f32 atomics (2 addends/site) ----
#pragma unroll
    for (int mg = 0; mg < 4; mg++) {
#pragma unroll
        for (int nt = 0; nt < 4; nt++) {
#pragma unroll
            for (int r = 0; r < 8; r++) {
                int m = mg * 16 + half * 8 + r;
                int h = w * 64 + nt * 16 + ln;
                atomicAdd(out + (size_t)stok[m] * HH + h, sp[m] * acc2[mg][nt][r]);
            }
        }
    }
}

// ---------------------------------------------------------------------------
extern "C" void kernel_launch(void* const* d_in, const int* in_sizes, int n_in,
                              void* d_out, int out_size, void* d_ws, size_t ws_size,
                              hipStream_t stream) {
    const float* x      = (const float*)d_in[0];   // [B,S,H]
    const float* logits = (const float*)d_in[1];   // [B,S,E]
    const float* w1     = (const float*)d_in[2];   // [E,H,F]
    const float* w2     = (const float*)d_in[3];   // [E,F,H]
    char*  ws  = (char*)d_ws;
    float* out = (float*)d_out;

    // 1) weights -> bf16 transposed (every call; no cached state)
    int tTiles = EE * (HH / 32) * (FF / 32) * 2;   // w1 + w2
    transpose_convert_kernel<<<tTiles, 256, 0, stream>>>(w1, w2, ws);

    // 2) routing
    router_kernel<<<(TT + 255) / 256, 256, 0, stream>>>(logits, ws);

    // 3) deterministic per-expert compaction (1 block, wave per expert)
    compact_kernel<<<1, 256, 0, stream>>>(ws);

    // 4) zero output
    int n4 = (TT * HH) / 4;
    zero_out_kernel<<<4096, 256, 0, stream>>>((float4*)out, n4);

    // 5) fused MoE FFN (dynamic LDS ~166 KB; CDNA5 WGP has 320 KB)
    size_t smem = (size_t)M_TILE * XP * 2 + (size_t)M_TILE * AP * 2
                + (size_t)M_TILE * 4 + (size_t)M_TILE * 4;
    moe_ffn_kernel<<<EE * (TT / M_TILE), BLOCK, smem, stream>>>(x, ws, out);
}
